// MTREncoder_49323404427931
// MI455X (gfx1250) — matrix operand order repaired and checked by reading, not compiled, not run on hardware
//
#include <hip/hip_runtime.h>
#include <math.h>

// ---------------------------------------------------------------------------
// MTR encoder for gfx1250 (MI455X).
//  * All GEMMs: v_wmma_f32_16x16x32_bf16, 128x128x32 block tiles, 8 waves,
//    8 WMMA per wave per K-step.
//  * A and W pre-converted to bf16 (W transposed to [N][K]); tile staging is a
//    raw copy via global_load_async_to_lds_b128 into double-buffered LDS
//    (ASYNCcnt pipeline overlapping DMA with WMMA).
//  * Kernel templated on HAS_A2 so the common path has branch-free staging.
//  * Wq/Wk fused into one N=512 projection GEMM (A staged once).
//  * Bias / ReLU / row-mask / residual / fused atomic max-pool / row-remap /
//    bf16-or-f32 output in the GEMM epilogue.
//  * Shapes guaranteed: M % 128 == 0, N % 128 == 0, K % 32 == 0.
// ---------------------------------------------------------------------------

typedef __attribute__((ext_vector_type(16))) __bf16 bf16x16;
typedef __attribute__((ext_vector_type(8)))  float  floatx8;

union FragAB {
  bf16x16 v;
  uint4 q[2];
};

__device__ __forceinline__ unsigned short f2bf(float f) {
  union { float f; unsigned int u; } c; c.f = f;
  unsigned int u = c.u;
  return (unsigned short)((u + 0x7FFFu + ((u >> 16) & 1u)) >> 16); // RNE
}

__device__ __forceinline__ void async_copy16(unsigned ldsoff, const void* g) {
  // raw 16B global -> LDS copy, tracked by ASYNCcnt
  asm volatile("global_load_async_to_lds_b128 %0, %1, off" :: "v"(ldsoff), "v"(g) : "memory");
}
__device__ __forceinline__ void wait_async0() {
  asm volatile("s_wait_asynccnt 0x0" ::: "memory");
}

#define BM 128
#define BN 128
#define BK 32
#define ROWH 40                 // halves per LDS tile row (32 data + 8 pad)
#define ATILE_H (BM * ROWH)     // 5120 halves
#define BUF_H   (2 * ATILE_H)   // A tile + W tile per buffer (10240 halves)

// C = epilogue( A[M,Kd](bf16) @ W[Kd,N] ), W pre-transposed bf16 WT[N][Kd].
// HAS_A2: A col k >= k_split comes from A2[(r/a2g)*(Kd-k_split) + k-k_split].
template <bool HAS_A2>
__global__ __launch_bounds__(256) void gemm_bf16_kernel(
    const unsigned short* __restrict__ A, const unsigned short* __restrict__ A2,
    int k_split, int a2_group,
    const unsigned short* __restrict__ WT, const float* __restrict__ bias,
    void* __restrict__ Cout, int c_bf16,
    const float* __restrict__ residual,
    const float* __restrict__ rowmask, float* __restrict__ pool, int pool_group,
    int M, int Kd, int N, int act,
    int out_group, int out_base, int out_stride)
{
  __shared__ __attribute__((aligned(16))) unsigned short smem[2 * BUF_H];
  const int tid  = threadIdx.x;
  const int lane = tid & 31;
  const int wave = tid >> 5;
  const int bm = blockIdx.x * BM;
  const int bn = blockIdx.y * BN;
  const int mw = wave & 3;      // 2 M-subtiles each
  const int nw = wave >> 2;     // 4 N-subtiles each
  const unsigned lds_base = (unsigned)(size_t)&smem[0];

  // staging geometry: 512 x 16B chunks per tile; this thread copies rows
  // srow and srow+64 at half-column scol (for both the A and W tiles).
  const int srow = tid >> 2;
  const int scol = (tid & 3) * 8;                       // halves
  const unsigned loff0 = lds_base + srow * (ROWH * 2) + (tid & 3) * 16;

  floatx8 acc[2][4] = {};
  const int nK = Kd / BK;

  auto stage = [&](int buf, int kt) {
    const int k0 = kt * BK;
    const unsigned bb = (unsigned)buf * (BUF_H * 2);
#pragma unroll
    for (int j = 0; j < 2; ++j) {                       // A tile
      int r = srow + 64 * j;
      unsigned loff = loff0 + bb + j * (64 * ROWH * 2);
      const unsigned short* g;
      if (!HAS_A2 || k0 < k_split)
        g = A + (size_t)(bm + r) * k_split + (k0 + scol);
      else
        g = A2 + (size_t)((bm + r) / a2_group) * (Kd - k_split)
               + (k0 - k_split + scol);
      async_copy16(loff, g);
    }
#pragma unroll
    for (int j = 0; j < 2; ++j) {                       // W tile
      int r = srow + 64 * j;
      unsigned loff = loff0 + bb + (ATILE_H * 2) + j * (64 * ROWH * 2);
      async_copy16(loff, WT + (size_t)(bn + r) * Kd + (k0 + scol));
    }
  };

  stage(0, 0);
  wait_async0();
  __syncthreads();

  for (int kt = 0; kt < nK; ++kt) {
    const int buf = kt & 1;
    if (kt + 1 < nK) stage(buf ^ 1, kt + 1);            // overlap DMA + WMMA

    const unsigned short* As = &smem[buf * BUF_H];
    const unsigned short* Ws = &smem[buf * BUF_H + ATILE_H];
    const int kA = (lane < 16) ? 0 : 8;                 // A: K 0-7/16-23 | 8-15/24-31
    const int kB = (lane < 16) ? 0 : 8;                 // B: K 0-7/16-23 | 8-15/24-31

    FragAB fa[2], fb[4];
#pragma unroll
    for (int im = 0; im < 2; ++im) {
      int mrow = (mw * 2 + im) * 16 + (lane & 15);
      fa[im].q[0] = *(const uint4*)(As + mrow * ROWH + kA);
      fa[im].q[1] = *(const uint4*)(As + mrow * ROWH + kA + 16);
    }
#pragma unroll
    for (int jn = 0; jn < 4; ++jn) {
      int nrow = (nw * 4 + jn) * 16 + (lane & 15);
      fb[jn].q[0] = *(const uint4*)(Ws + nrow * ROWH + kB);
      fb[jn].q[1] = *(const uint4*)(Ws + nrow * ROWH + kB + 16);
    }
#pragma unroll
    for (int im = 0; im < 2; ++im)
#pragma unroll
      for (int jn = 0; jn < 4; ++jn)
        acc[im][jn] = __builtin_amdgcn_wmma_f32_16x16x32_bf16(
            false, fa[im].v, false, fb[jn].v, (short)0, acc[im][jn], false, false);

    wait_async0();
    __syncthreads();
  }

  // epilogue: C/D layout vgpr r -> (M=r | M=8+r), N = lane&15
#pragma unroll
  for (int im = 0; im < 2; ++im) {
#pragma unroll
    for (int jn = 0; jn < 4; ++jn) {
      floatx8 a = acc[im][jn];
      int col = bn + (nw * 4 + jn) * 16 + (lane & 15);
      float bsv = bias ? bias[col] : 0.f;
#pragma unroll
      for (int r = 0; r < 8; ++r) {
        int row = bm + (mw * 2 + im) * 16 + r + ((lane < 16) ? 0 : 8);
        float v = a[r] + bsv;
        if (act && v < 0.f) v = 0.f;
        if (rowmask)  v *= rowmask[row];
        if (residual) v += residual[(size_t)row * N + col];
        if (Cout) {
          size_t crow = (size_t)row;
          if (out_group > 0)
            crow = (size_t)(row / out_group) * out_stride + out_base + (row % out_group);
          if (c_bf16) ((unsigned short*)Cout)[crow * N + col] = f2bf(v);
          else        ((float*)Cout)[crow * N + col] = v;
        }
        if (pool)
          __hip_atomic_fetch_max(&pool[(size_t)(row / pool_group) * N + col], v,
                                 __ATOMIC_RELAXED, __HIP_MEMORY_SCOPE_AGENT);
      }
    }
  }
}

// ---------------------------------------------------------------------------
// conversion / prep kernels
// ---------------------------------------------------------------------------

// W[K,N] f32  ->  out[N][Kpad] bf16 (transposed, zero-padded K)
__global__ void wt_convert_kernel(const float* __restrict__ W,
                                  unsigned short* __restrict__ out,
                                  int K, int N, int Kpad) {
  int i = blockIdx.x * blockDim.x + threadIdx.x;
  if (i >= N * Kpad) return;
  int n = i / Kpad, k = i % Kpad;
  out[i] = (k < K) ? f2bf(W[(size_t)k * N + n]) : (unsigned short)0;
}

__global__ void f32_to_bf16_kernel(const float* __restrict__ in,
                                   unsigned short* __restrict__ out, int n) {
  int i = blockIdx.x * blockDim.x + threadIdx.x;
  if (i < n) out[i] = f2bf(in[i]);
}

__global__ void concat2_kernel(const float* __restrict__ a, const float* __restrict__ b,
                               float* __restrict__ o, int n) {
  int i = blockIdx.x * blockDim.x + threadIdx.x;
  if (i < n) o[i] = a[i];
  else if (i < 2 * n) o[i] = b[i - n];
}

// obj_in = [trajs(29) | mask | 0 0] as bf16, rows padded to 32 cols
__global__ void build_obj_in_kernel(const float* __restrict__ trajs,
                                    const unsigned char* __restrict__ mask,
                                    unsigned short* __restrict__ out, int n) {
  int i = blockIdx.x * blockDim.x + threadIdx.x;
  if (i >= n) return;
  int p = i >> 5, c = i & 31;
  float f = 0.f;
  if (c < 29)       f = trajs[(size_t)p * 29 + c];
  else if (c == 29) f = mask[p] ? 1.f : 0.f;
  out[i] = f2bf(f);
}

// f32 [rows][cin] -> bf16 [rows][cpad] zero-padded
__global__ void pad_convert_kernel(const float* __restrict__ in,
                                   unsigned short* __restrict__ out,
                                   int rows, int cin, int cpad) {
  int i = blockIdx.x * blockDim.x + threadIdx.x;
  if (i >= rows * cpad) return;
  int r = i / cpad, c = i % cpad;
  out[i] = (c < cin) ? f2bf(in[(size_t)r * cin + c]) : (unsigned short)0;
}

__global__ void mask_to_float_kernel(const unsigned char* __restrict__ in,
                                     float* __restrict__ out, int n) {
  int i = blockIdx.x * blockDim.x + threadIdx.x;
  if (i < n) out[i] = in[i] ? 1.f : 0.f;
}

__global__ void validity_kernel(const unsigned char* __restrict__ m,
                                float* __restrict__ v, int groups, int glen) {
  int i = blockIdx.x * blockDim.x + threadIdx.x;
  if (i >= groups) return;
  int s = 0;
  for (int t = 0; t < glen; ++t) s += m[(size_t)i * glen + t] ? 1 : 0;
  v[i] = (s > 0) ? 1.f : 0.f;
}

__global__ void pos_kernel(const float* __restrict__ lastpos,
                           const float* __restrict__ center,
                           const float* __restrict__ vobj,
                           const float* __restrict__ vmap,
                           float* __restrict__ posm, float* __restrict__ mtok) {
  int t = blockIdx.x * blockDim.x + threadIdx.x;
  if (t >= 16 * 576) return;
  int b = t / 576, j = t % 576;
  float m; const float* p;
  if (j < 64) { m = vobj[b * 64 + j];         p = lastpos + ((size_t)b * 64 + j) * 3; }
  else        { m = vmap[b * 512 + (j - 64)]; p = center  + ((size_t)b * 512 + (j - 64)) * 3; }
  mtok[t] = m;
  posm[(size_t)t * 3 + 0] = p[0] * m;
  posm[(size_t)t * 3 + 1] = p[1] * m;
  posm[(size_t)t * 3 + 2] = p[2] * m;
}

__global__ void knn_kernel(const float* __restrict__ posm,
                           const float* __restrict__ mtok,
                           int* __restrict__ nidx, float* __restrict__ nval) {
  int t = blockIdx.x * blockDim.x + threadIdx.x;
  if (t >= 16 * 576) return;
  int b = t / 576, base = b * 576;
  float qx = posm[(size_t)t * 3], qy = posm[(size_t)t * 3 + 1], qz = posm[(size_t)t * 3 + 2];
  float bd[16]; int bi[16];
  for (int i = 0; i < 16; ++i) { bd[i] = 3e38f; bi[i] = base; }
  for (int j = 0; j < 576; ++j) {
    const float* pk = posm + (size_t)(base + j) * 3;
    float dx = qx - pk[0], dy = qy - pk[1], dz = qz - pk[2];
    float d2 = dx * dx + dy * dy + dz * dz;
    if (mtok[base + j] <= 0.f) d2 = 1e9f;     // BIG for invalid keys
    if (d2 < bd[15]) {                        // strict < == top_k tie order
      int p = 15;
      while (p > 0 && bd[p - 1] > d2) { bd[p] = bd[p - 1]; bi[p] = bi[p - 1]; --p; }
      bd[p] = d2; bi[p] = base + j;           // global token index
    }
  }
  float qm = mtok[t];
  for (int i = 0; i < 16; ++i) {
    nidx[(size_t)t * 16 + i] = bi[i];
    nval[(size_t)t * 16 + i] = (bd[i] < 5e8f && qm > 0.f) ? 1.f : 0.f;
  }
}

__global__ void pe_kernel(const float* __restrict__ posm,
                          const float* __restrict__ mtok, float* __restrict__ pe) {
  int idx = blockIdx.x * blockDim.x + threadIdx.x;
  if (idx >= 16 * 576 * 256) return;
  int t = idx >> 8, c = idx & 255;
  float coord = (c < 128) ? posm[(size_t)t * 3 + 1] : posm[(size_t)t * 3 + 0]; // y then x
  int i = c & 127;
  float dim_t = powf(10000.f, (float)(2 * (i / 2)) / 128.f);
  float p = coord * 6.283185307179586f / dim_t;
  float val = (i & 1) ? cosf(p) : sinf(p);
  pe[idx] = val * mtok[t];
}

__global__ void add_bf_kernel(const float* __restrict__ a, const float* __restrict__ b,
                              unsigned short* __restrict__ o, int n) {
  int i = blockIdx.x * blockDim.x + threadIdx.x;
  if (i < n) o[i] = f2bf(a[i] + b[i]);
}

// one block (256 threads) per token: H=8 heads, K=16 neighbors, dh=32.
// qk packed [NT][512] (q = cols 0-255, k = cols 256-511).
__global__ __launch_bounds__(256) void attn_kernel(
    const float* __restrict__ qk, const float* __restrict__ v,
    const int* __restrict__ nidx, const float* __restrict__ nval,
    unsigned short* __restrict__ outbf) {
  __shared__ float s[8][16];
  __shared__ int   sidx[16];
  __shared__ float sval[16];
  int t = blockIdx.x, tid = threadIdx.x;
  if (tid < 16) { sidx[tid] = nidx[(size_t)t * 16 + tid]; sval[tid] = nval[(size_t)t * 16 + tid]; }
  __syncthreads();
  if (tid < 128) {
    int h = tid >> 4, kn = tid & 15;
    const float* qp = qk + (size_t)t * 512 + h * 32;
    const float* kp = qk + (size_t)sidx[kn] * 512 + 256 + h * 32;
    float acc = 0.f;
    for (int d = 0; d < 32; ++d) acc += qp[d] * kp[d];
    acc *= 0.17677669529663687f;  // 1/sqrt(32)
    s[h][kn] = (sval[kn] > 0.f) ? acc : -1e9f;
  }
  __syncthreads();
  if (tid < 8) {
    int h = tid;
    float mx = -3.4e38f;
    for (int i = 0; i < 16; ++i) mx = fmaxf(mx, s[h][i]);
    float e[16]; float sm = 0.f;
    for (int i = 0; i < 16; ++i) { e[i] = __expf(s[h][i] - mx); sm += e[i]; }
    float inv = 1.f / sm;
    for (int i = 0; i < 16; ++i) s[h][i] = e[i] * inv;
  }
  __syncthreads();
  {
    int h = tid >> 5, d = tid & 31;
    float acc = 0.f;
    for (int kn = 0; kn < 16; ++kn)
      acc += s[h][kn] * v[(size_t)sidx[kn] * 256 + h * 32 + d];
    outbf[(size_t)t * 256 + h * 32 + d] = f2bf(acc);
  }
}

// wave-per-row layernorm over 256 cols; writes f32 and bf16, optional *mtok
__global__ __launch_bounds__(256) void ln_kernel(const float* __restrict__ in,
                                                 const float* __restrict__ g,
                                                 const float* __restrict__ be,
                                                 const float* __restrict__ mtok,
                                                 float* __restrict__ out,
                                                 unsigned short* __restrict__ outbf,
                                                 int rows) {
  int wave = threadIdx.x >> 5, lane = threadIdx.x & 31;
  int row = blockIdx.x * 8 + wave;
  if (row >= rows) return;
  const float* p = in + (size_t)row * 256;
  float v[8]; float s = 0.f;
  for (int j = 0; j < 8; ++j) { v[j] = p[lane + 32 * j]; s += v[j]; }
  for (int m = 16; m >= 1; m >>= 1) s += __shfl_xor(s, m, 32);
  float mu = s * (1.f / 256.f);
  float s2 = 0.f;
  for (int j = 0; j < 8; ++j) { float d = v[j] - mu; s2 += d * d; }
  for (int m = 16; m >= 1; m >>= 1) s2 += __shfl_xor(s2, m, 32);
  float inv = rsqrtf(s2 * (1.f / 256.f) + 1e-5f);
  float mm = mtok ? mtok[row] : 1.f;
  for (int j = 0; j < 8; ++j) {
    int c = lane + 32 * j;
    float o = ((v[j] - mu) * inv * g[c] + be[c]) * mm;
    out[(size_t)row * 256 + c] = o;
    if (outbf) outbf[(size_t)row * 256 + c] = f2bf(o);
  }
}

__global__ void out_kernel(const float* __restrict__ x, const float* __restrict__ vobj,
                           const float* __restrict__ vmap, const float* __restrict__ lastpos,
                           const float* __restrict__ center, const int* __restrict__ track,
                           float* __restrict__ out) {
  int i = blockIdx.x * blockDim.x + threadIdx.x;
  const int S0 = 262144, S1 = S0 + 1024, S2 = S1 + 3072, S3 = S2 + 2097152,
            S4 = S3 + 8192, S5 = S4 + 24576, S6 = S5 + 4096;
  if (i >= S6) return;
  if (i < S0) {                        // obj_out [16,64,256]
    int b = i >> 14, r = (i >> 8) & 63, c = i & 255;
    out[i] = x[((size_t)b * 576 + r) * 256 + c];
  } else if (i < S1) out[i] = vobj[i - S0];
  else if (i < S2)   out[i] = lastpos[i - S1];
  else if (i < S3) {                   // map_out [16,512,256]
    int kk = i - S2; int b = kk / (512 * 256); int r = (kk >> 8) % 512; int c = kk & 255;
    out[i] = x[((size_t)b * 576 + 64 + r) * 256 + c];
  } else if (i < S4) out[i] = vmap[i - S3];
  else if (i < S5)   out[i] = center[i - S4];
  else {                               // center token feature
    int kk = i - S5; int b = kk >> 8; int c = kk & 255;
    out[i] = x[((size_t)b * 576 + track[b]) * 256 + c];
  }
}

// ---------------------------------------------------------------------------
// host side
// ---------------------------------------------------------------------------
static inline int cdiv(int a, int b) { return (a + b - 1) / b; }

static void launch_gemm(hipStream_t st, const unsigned short* A, const unsigned short* A2,
                        int ksplit, int a2g, const unsigned short* WT, const float* bias,
                        void* C, int c_bf16, const float* res, const float* rmask,
                        float* pool, int pg, int M, int K, int N, int act,
                        int og = 0, int obase = 0, int ostride = 0) {
  dim3 grid(M / BM, N / BN);
  if (A2)
    gemm_bf16_kernel<true><<<grid, 256, 0, st>>>(A, A2, ksplit, a2g, WT, bias, C, c_bf16,
                                                 res, rmask, pool, pg, M, K, N, act,
                                                 og, obase, ostride);
  else
    gemm_bf16_kernel<false><<<grid, 256, 0, st>>>(A, A2, ksplit, a2g, WT, bias, C, c_bf16,
                                                  res, rmask, pool, pg, M, K, N, act,
                                                  og, obase, ostride);
}

extern "C" void kernel_launch(void* const* d_in, const int* in_sizes, int n_in,
                              void* d_out, int out_size, void* d_ws, size_t ws_size,
                              hipStream_t stream) {
  (void)in_sizes; (void)n_in; (void)out_size; (void)ws_size;
  const float*         obj_trajs = (const float*)d_in[0];
  const unsigned char* omask     = (const unsigned char*)d_in[1];
  const float*         map_poly  = (const float*)d_in[2];
  const unsigned char* mmask     = (const unsigned char*)d_in[3];
  const float*         map_ctr   = (const float*)d_in[4];
  const float*         last_pos  = (const float*)d_in[5];
  const int*           track     = (const int*)d_in[6];
  const float* agW1 = (const float*)d_in[7];  const float* agW2 = (const float*)d_in[8];
  const float* agW3 = (const float*)d_in[9];  const float* agb1 = (const float*)d_in[10];
  const float* agb2 = (const float*)d_in[11]; const float* agb3 = (const float*)d_in[12];
  auto LP = [&](int l, int j) { return (const float*)d_in[13 + l * 16 + j]; };
  const float* mpW1 = (const float*)d_in[77]; const float* mpW2 = (const float*)d_in[78];
  const float* mpW3 = (const float*)d_in[79]; const float* mpb1 = (const float*)d_in[80];
  const float* mpb2 = (const float*)d_in[81]; const float* mpb3 = (const float*)d_in[82];

  char* base = (char*)d_ws;
  size_t off = 0;
  auto alloc = [&](size_t bytes) { off = (off + 255) & ~(size_t)255; size_t o = off; off += bytes; return o; };
  auto F  = [&](size_t n) { return (float*)(base + alloc(n * 4)); };
  auto H  = [&](size_t n) { return (unsigned short*)(base + alloc(n * 2)); };

  // ---- f32 buffers --------------------------------------------------------
  float* X    = F(9216 * 256);
  float* PE   = F(9216 * 256);
  float* MTOK = F(9216);
  float* POSM = F(9216 * 3);
  int*   NIDX = (int*)(base + alloc(9216 * 16 * 4));
  float* NVAL = F(9216 * 16);
  float* VOBJ = F(1024);
  float* VMAP = F(8192);
  float* PMOBJ = F(11264);
  float* PMMAP = F(163840);
  float* QKF = F(9216 * 512);             // packed q|k
  float* VF  = F(9216 * 256);
  float* TMP = F(9216 * 256); float* XAF = F(9216 * 256);
  float* PLOBJ = F(1024 * 256);  float* FOBJ = F(1024 * 256);
  float* PLMAP = F(8192 * 256);  float* FMAP = F(8192 * 256);
  // ---- bf16 buffers -------------------------------------------------------
  unsigned short* OBJIN = H(11264 * 32);
  unsigned short* MAPIN = H(163840 * 32);
  unsigned short* HOBJ  = H(11264 * 256);
  unsigned short* HMAPC = H(20480 * 256);
  unsigned short* PLOBJB = H(1024 * 256); unsigned short* FOBJB = H(1024 * 256);
  unsigned short* PLMAPB = H(8192 * 256); unsigned short* FMAPB = H(8192 * 256);
  unsigned short* XB   = H(9216 * 256);
  unsigned short* QKIN = H(9216 * 256);
  unsigned short* AOB  = H(9216 * 256);
  unsigned short* XAB  = H(9216 * 256);
  unsigned short* HIDB = H(9216 * 1024);
  // ---- transposed bf16 weights -------------------------------------------
  unsigned short* AG1T = H(256 * 32);   unsigned short* AG2T = H(256 * 512);
  unsigned short* AG3T = H(256 * 256);
  unsigned short* MP1T = H(256 * 32);   unsigned short* MP2T = H(256 * 512);
  unsigned short* MP3T = H(256 * 256);
  unsigned short* LQK[4]; unsigned short* LV[4]; unsigned short* LO[4];
  unsigned short* LF1[4]; unsigned short* LF2[4];
  float* QKB[4];
  for (int l = 0; l < 4; ++l) {
    LQK[l] = H(512 * 256);                // Wq rows 0-255, Wk rows 256-511
    LV[l]  = H(256 * 256);
    LO[l]  = H(256 * 256);
    LF1[l] = H(1024 * 256);               // Wf1 [256,1024] -> [1024][256]
    LF2[l] = H(256 * 1024);               // Wf2 [1024,256] -> [256][1024]
    QKB[l] = F(512);                      // bq | bk
  }

  auto wconv = [&](const float* W, unsigned short* o, int K, int N, int Kpad) {
    int n = N * Kpad;
    wt_convert_kernel<<<cdiv(n, 256), 256, 0, stream>>>(W, o, K, N, Kpad);
  };
  wconv(agW1, AG1T, 30, 256, 32);  wconv(agW2, AG2T, 512, 256, 512);
  wconv(agW3, AG3T, 256, 256, 256);
  wconv(mpW1, MP1T, 9, 256, 32);   wconv(mpW2, MP2T, 512, 256, 512);
  wconv(mpW3, MP3T, 256, 256, 256);
  for (int l = 0; l < 4; ++l) {
    wconv(LP(l, 4), LQK[l],              256, 256, 256);   // Wq
    wconv(LP(l, 2), LQK[l] + 256 * 256,  256, 256, 256);   // Wk
    wconv(LP(l, 5), LV[l],  256, 256, 256);                // Wv
    wconv(LP(l, 3), LO[l],  256, 256, 256);                // Wo
    wconv(LP(l, 0), LF1[l], 256, 1024, 256);               // Wf1
    wconv(LP(l, 1), LF2[l], 1024, 256, 1024);              // Wf2
    concat2_kernel<<<cdiv(512, 256), 256, 0, stream>>>(LP(l, 12), LP(l, 10), QKB[l], 256);
  }

  // ---- inputs / masks -----------------------------------------------------
  build_obj_in_kernel<<<cdiv(11264 * 32, 256), 256, 0, stream>>>(obj_trajs, omask, OBJIN, 11264 * 32);
  pad_convert_kernel<<<cdiv(163840 * 32, 256), 256, 0, stream>>>(map_poly, MAPIN, 163840, 9, 32);
  mask_to_float_kernel<<<cdiv(11264, 256), 256, 0, stream>>>(omask, PMOBJ, 11264);
  mask_to_float_kernel<<<cdiv(163840, 256), 256, 0, stream>>>(mmask, PMMAP, 163840);
  validity_kernel<<<cdiv(1024, 256), 256, 0, stream>>>(omask, VOBJ, 1024, 11);
  validity_kernel<<<cdiv(8192, 256), 256, 0, stream>>>(mmask, VMAP, 8192, 20);
  hipMemsetAsync(PLOBJ, 0, 1024 * 256 * 4, stream);
  hipMemsetAsync(FOBJ,  0, 1024 * 256 * 4, stream);
  hipMemsetAsync(PLMAP, 0, 8192 * 256 * 4, stream);
  hipMemsetAsync(FMAP,  0, 8192 * 256 * 4, stream);

  // ---- agent polyline encoder --------------------------------------------
  launch_gemm(stream, OBJIN, nullptr, 32, 1, AG1T, agb1, HOBJ, 1, nullptr,
              PMOBJ, PLOBJ, 11, 11264, 32, 256, 1);
  f32_to_bf16_kernel<<<cdiv(262144, 256), 256, 0, stream>>>(PLOBJ, PLOBJB, 262144);
  launch_gemm(stream, HOBJ, PLOBJB, 256, 11, AG2T, agb2, nullptr, 0, nullptr,
              PMOBJ, FOBJ, 11, 11264, 512, 256, 1);
  f32_to_bf16_kernel<<<cdiv(262144, 256), 256, 0, stream>>>(FOBJ, FOBJB, 262144);
  launch_gemm(stream, FOBJB, nullptr, 256, 1, AG3T, agb3, X, 0, nullptr,
              VOBJ, nullptr, 1, 1024, 256, 256, 0, 64, 0, 576);

  // ---- map polyline encoder (chunks of 2 batches = 20480 points) ----------
  for (int c = 0; c < 8; ++c) {
    const unsigned short* Ain = MAPIN + (size_t)c * 20480 * 32;
    float* pl = PLMAP + (size_t)c * 262144;
    float* fm = FMAP + (size_t)c * 262144;
    unsigned short* plb = PLMAPB + (size_t)c * 262144;
    const float* pm = PMMAP + (size_t)c * 20480;
    launch_gemm(stream, Ain, nullptr, 32, 1, MP1T, mpb1, HMAPC, 1, nullptr,
                pm, pl, 20, 20480, 32, 256, 1);
    f32_to_bf16_kernel<<<cdiv(262144, 256), 256, 0, stream>>>(pl, plb, 262144);
    launch_gemm(stream, HMAPC, plb, 256, 20, MP2T, mpb2, nullptr, 0, nullptr,
                pm, fm, 20, 20480, 512, 256, 1);
  }
  f32_to_bf16_kernel<<<cdiv(2097152, 256), 256, 0, stream>>>(FMAP, FMAPB, 2097152);
  launch_gemm(stream, FMAPB, nullptr, 256, 1, MP3T, mpb3, X, 0, nullptr,
              VMAP, nullptr, 1, 8192, 256, 256, 0, 512, 64, 576);

  // ---- positions / knn / positional embedding -----------------------------
  pos_kernel<<<cdiv(9216, 256), 256, 0, stream>>>(last_pos, map_ctr, VOBJ, VMAP, POSM, MTOK);
  knn_kernel<<<cdiv(9216, 256), 256, 0, stream>>>(POSM, MTOK, NIDX, NVAL);
  pe_kernel<<<cdiv(9216 * 256, 256), 256, 0, stream>>>(POSM, MTOK, PE);
  f32_to_bf16_kernel<<<cdiv(9216 * 256, 256), 256, 0, stream>>>(X, XB, 9216 * 256);

  // ---- transformer layers -------------------------------------------------
  const int NT = 9216, NE = 9216 * 256;
  for (int l = 0; l < 4; ++l) {
    const float *be1 = LP(l, 6), *be2 = LP(l, 7), *bf1 = LP(l, 8), *bf2 = LP(l, 9),
                *bo = LP(l, 11), *bv = LP(l, 13), *g1 = LP(l, 14), *g2 = LP(l, 15);
    add_bf_kernel<<<cdiv(NE, 256), 256, 0, stream>>>(X, PE, QKIN, NE);
    launch_gemm(stream, QKIN, nullptr, 256, 1, LQK[l], QKB[l], QKF, 0, nullptr,
                nullptr, nullptr, 1, NT, 256, 512, 0);          // fused q|k
    launch_gemm(stream, XB, nullptr, 256, 1, LV[l], bv, VF, 0, nullptr,
                nullptr, nullptr, 1, NT, 256, 256, 0);
    attn_kernel<<<NT, 256, 0, stream>>>(QKF, VF, NIDX, NVAL, AOB);
    launch_gemm(stream, AOB, nullptr, 256, 1, LO[l], bo, TMP, 0, X,
                nullptr, nullptr, 1, NT, 256, 256, 0);
    ln_kernel<<<cdiv(NT, 8), 256, 0, stream>>>(TMP, g1, be1, nullptr, XAF, XAB, NT);
    launch_gemm(stream, XAB, nullptr, 256, 1, LF1[l], bf1, HIDB, 1, nullptr,
                nullptr, nullptr, 1, NT, 256, 1024, 1);
    launch_gemm(stream, HIDB, nullptr, 1024, 1, LF2[l], bf2, TMP, 0, XAF,
                nullptr, nullptr, 1, NT, 1024, 256, 0);
    ln_kernel<<<cdiv(NT, 8), 256, 0, stream>>>(TMP, g2, be2, MTOK, X, XB, NT);
  }

  // ---- outputs ------------------------------------------------------------
  out_kernel<<<cdiv(2400256, 256), 256, 0, stream>>>(X, VOBJ, VMAP, last_pos, map_ctr,
                                                     track, (float*)d_out);
}
